// Attention_67654324847232
// MI455X (gfx1250) — matrix operand order repaired and checked by reading, not compiled, not run on hardware
//
#include <hip/hip_runtime.h>

#define NQQ 8192
#define NKK 8192
#define DD  256
#define NKW (NKK / 32)   // mask words per row

// scale (1/sqrt(256)) folded with log2(e) so we can use exp2
#define SC_LOG2E 0.09016844005556021f
#define NEG_BIG  (-1e30f)

typedef __attribute__((ext_vector_type(16))) __bf16 v16bf;
typedef __attribute__((ext_vector_type(8)))  float  v8f;
typedef __attribute__((ext_vector_type(4)))  unsigned int u32x4;
typedef __attribute__((ext_vector_type(8)))  int          i32x8;
typedef __attribute__((ext_vector_type(4)))  int          i32x4;

union FragBF {
    v16bf v;
    struct { float4 lo, hi; } f;
};

// ---------------------------------------------------------------------------
// Pre-pass 1: fp32 -> bf16; K,Q row-major, V transposed to [D][NK]
// ---------------------------------------------------------------------------
__global__ void convert_kernel(const float* __restrict__ Kg,
                               const float* __restrict__ Vg,
                               const float* __restrict__ Qg,
                               __bf16* __restrict__ Kbf,
                               __bf16* __restrict__ Vtg,
                               __bf16* __restrict__ Qbf) {
    const size_t i = (size_t)blockIdx.x * blockDim.x + threadIdx.x;  // over NK*D
    const int key = (int)(i >> 8);
    const int d   = (int)(i & 255);
    Kbf[i] = (__bf16)Kg[i];
    Qbf[i] = (__bf16)Qg[i];
    Vtg[(size_t)d * NKK + key] = (__bf16)Vg[i];
}

// ---------------------------------------------------------------------------
// Pre-pass 2: pack bool mask bytes into u32 bitmask words (bit j = key j)
// ---------------------------------------------------------------------------
__global__ void maskpack_kernel(const unsigned char* __restrict__ Mg,
                                unsigned* __restrict__ Mw) {
    const size_t w = (size_t)blockIdx.x * blockDim.x + threadIdx.x;  // over NQ*NKW
    const unsigned* src = (const unsigned*)(Mg + w * 32);
    unsigned bits = 0;
    #pragma unroll
    for (int u = 0; u < 8; ++u) {
        const unsigned x = src[u];
        #pragma unroll
        for (int k = 0; k < 4; ++k)
            bits |= ((x >> (8 * k)) & 1u) << (4 * u + k);
    }
    Mw[w] = bits;
}

// ---------------------------------------------------------------------------
// TDM: build a 2D-tile D# and issue tensor_load_to_lds (wave-level, EXEC-free)
// ---------------------------------------------------------------------------
__device__ __forceinline__ void tdm_load_2d(unsigned lds_off,
                                            const void* gptr,
                                            unsigned tensor_d0, unsigned tensor_d1,
                                            unsigned long long stride0,
                                            unsigned tile_d0, unsigned tile_d1,
                                            unsigned data_size_code,     // 0=1B,1=2B,2=4B
                                            unsigned pad_en,
                                            unsigned pad_interval_code,  // (1<<c)*8B
                                            unsigned pad_amount_code) {  // (c+1) dwords
    const unsigned long long ga = (unsigned long long)(size_t)gptr;
    u32x4 g0;
    g0[0] = 1u;                                   // count=1, user descriptor
    g0[1] = lds_off;                              // LDS byte address
    g0[2] = (unsigned)(ga & 0xffffffffu);         // global_addr[31:0]
    g0[3] = (unsigned)((ga >> 32) & 0x01ffffffu)  // global_addr[56:32]
          | (2u << 30);                           // type = 2 ("image")
    i32x8 g1;
    g1[0] = (int)((data_size_code << 16)
          | (pad_en << 20)
          | (pad_interval_code << 22)
          | (pad_amount_code  << 25));
    g1[1] = (int)((tensor_d0 & 0xffffu) << 16);                       // dim0[15:0]
    g1[2] = (int)((tensor_d0 >> 16) | ((tensor_d1 & 0xffffu) << 16)); // dim0[31:16]|dim1[15:0]
    g1[3] = (int)((tensor_d1 >> 16) | (tile_d0 << 16));               // dim1[31:16]|tile0
    g1[4] = (int)(tile_d1 & 0xffffu);                                 // tile1, tile2=0
    g1[5] = (int)(unsigned)(stride0 & 0xffffffffu);
    g1[6] = (int)(unsigned)((stride0 >> 32) & 0xffffu);
    g1[7] = 0;
    const i32x4 z4 = {0, 0, 0, 0};
    const i32x8 z8 = {0, 0, 0, 0, 0, 0, 0, 0};
    __builtin_amdgcn_tensor_load_to_lds(g0, g1, z4, z4, z8, 0);
}

// ---------------------------------------------------------------------------
// Flash attention main kernel: 4 waves/block, 16 queries/wave, 32 keys/step
// ---------------------------------------------------------------------------
__launch_bounds__(128, 1)
__global__ void fa_wmma_kernel(const __bf16* __restrict__ Kbf,
                               const __bf16* __restrict__ Vtg,
                               const __bf16* __restrict__ Qbf,
                               const unsigned* __restrict__ Mwg,
                               float* __restrict__ Og) {
    // double-buffered TDM targets (row strides match D# padding)
    __shared__ __align__(32) __bf16 Kt[2][32][272];   // 32 keys x 256d, 544B rows
    __shared__ __align__(32) __bf16 Vt[2][256][48];   // 256d x 32 keys, 96B rows
    __shared__ __align__(16) unsigned Mw[2][64];      // 64 q-rows x 1 mask word
    __shared__ __align__(16) __bf16 Pb[4][16][40];    // per-wave P transpose scratch

    const int tid  = threadIdx.x;
    const int lane = tid & 31;
    const int wave = tid >> 5;
    const int l16  = lane & 15;
    const int sub  = lane >> 4;

    const int qbase0 = blockIdx.x * 64;
    const int qbase  = qbase0 + wave * 16;

    // ---- Load Q fragments (already bf16, exact A-layout: two 16B loads/chunk) ----
    v16bf qa[8];
    {
        const __bf16* qrow = Qbf + (size_t)(qbase + l16) * DD;
        #pragma unroll
        for (int kc = 0; kc < 8; ++kc) {
            const int d0 = kc * 32;
            FragBF a;
            a.f.lo = *(const float4*)(qrow + d0 + sub * 8);
            a.f.hi = *(const float4*)(qrow + d0 + 16 + sub * 8);
            qa[kc] = a.v;
        }
    }

    v8f o[16];
    {
        v8f z = {};
        #pragma unroll
        for (int c = 0; c < 16; ++c) o[c] = z;
    }
    float mrow[8], lrow[8];
    #pragma unroll
    for (int r = 0; r < 8; ++r) { mrow[r] = NEG_BIG; lrow[r] = 0.0f; }

    const unsigned lds_K = (unsigned)(size_t)&Kt[0][0][0];
    const unsigned lds_V = (unsigned)(size_t)&Vt[0][0][0];
    const unsigned lds_M = (unsigned)(size_t)&Mw[0][0];
    const unsigned szK = sizeof(Kt[0]);
    const unsigned szV = sizeof(Vt[0]);
    const unsigned szM = sizeof(Mw[0]);

    // Prologue: TDM stage of block kb=0 into buffer 0 (wave 0 only)
    if (wave == 0) {
        // K tile: 512B rows, pad +32B -> interval (1<<6)*8B (code 6), amount 8dw (code 7)
        tdm_load_2d(lds_K, Kbf, DD, NKK, DD, DD, 32, 1, 1, 6, 7);
        // Vt tile: 64B rows, pad +32B -> interval (1<<3)*8B (code 3), amount 8dw (code 7)
        tdm_load_2d(lds_V, Vtg, NKK, DD, NKK, 32, DD, 1, 1, 3, 7);
        // mask words: 64 rows x 1 u32, no padding
        tdm_load_2d(lds_M, Mwg + (size_t)qbase0 * NKW, NKW, NQQ, NKW, 1, 64, 2, 0, 0, 0);
    }

    int buf = 0;
    for (int kb = 0; kb < NKK; kb += 32) {
        if (wave == 0) __builtin_amdgcn_s_wait_tensorcnt(0);
        __syncthreads();   // tiles in `buf` ready; everyone done with `buf^1`

        // Kick off next block into the other buffer while we compute
        if (wave == 0 && (kb + 32) < NKK) {
            const int nb = buf ^ 1;
            const int kn = kb + 32;
            tdm_load_2d(lds_K + nb * szK, Kbf + (size_t)kn * DD, DD, NKK, DD, DD, 32, 1, 1, 6, 7);
            tdm_load_2d(lds_V + nb * szV, Vtg + kn, NKK, DD, NKK, 32, DD, 1, 1, 3, 7);
            tdm_load_2d(lds_M + nb * szM, Mwg + (size_t)qbase0 * NKW + (kn >> 5),
                        NKW, NQQ, NKW, 1, 64, 2, 0, 0, 0);
        }

        // ---- S = Q * K^T : two 16x16 key tiles, K-dim 256 in 8 chunks ----
        v8f s0 = {}, s1 = {};
        #pragma unroll
        for (int kc = 0; kc < 8; ++kc) {
            const int off = kc * 32 + sub * 8;
            FragBF b0, b1;
            b0.f.lo = *(const float4*)&Kt[buf][l16][off];
            b0.f.hi = *(const float4*)&Kt[buf][l16][off + 16];
            b1.f.lo = *(const float4*)&Kt[buf][16 + l16][off];
            b1.f.hi = *(const float4*)&Kt[buf][16 + l16][off + 16];
            s0 = __builtin_amdgcn_wmma_f32_16x16x32_bf16(false, qa[kc], false, b0.v,
                                                         (short)0, s0, false, false);
            s1 = __builtin_amdgcn_wmma_f32_16x16x32_bf16(false, qa[kc], false, b1.v,
                                                         (short)0, s1, false, false);
        }

        // ---- mask (bitmask words) + online softmax (exp2 domain) ----
        float alpha[8];
        int need = 0;
        #pragma unroll
        for (int r = 0; r < 8; ++r) {
            const unsigned w = Mw[buf][wave * 16 + r + 8 * sub] >> l16;
            float v0 = (w & 1u)         ? s0[r] * SC_LOG2E : NEG_BIG;
            float v1 = ((w >> 16) & 1u) ? s1[r] * SC_LOG2E : NEG_BIG;

            float t = fmaxf(v0, v1);
            t = fmaxf(t, __shfl_xor(t, 1));
            t = fmaxf(t, __shfl_xor(t, 2));
            t = fmaxf(t, __shfl_xor(t, 4));
            t = fmaxf(t, __shfl_xor(t, 8));

            const float mn = fmaxf(mrow[r], t);
            const float a  = exp2f(mrow[r] - mn);
            const float p0 = exp2f(v0 - mn);
            const float p1 = exp2f(v1 - mn);

            float rs = p0 + p1;
            rs += __shfl_xor(rs, 1);
            rs += __shfl_xor(rs, 2);
            rs += __shfl_xor(rs, 4);
            rs += __shfl_xor(rs, 8);

            lrow[r] = lrow[r] * a + rs;
            mrow[r] = mn;
            alpha[r] = a;
            need |= (a != 1.0f);
            s0[r] = p0;
            s1[r] = p1;
        }

        // rescale O only when some row max actually moved (rare after warmup)
        if (__any(need)) {
            #pragma unroll
            for (int c = 0; c < 16; ++c) {
                #pragma unroll
                for (int r = 0; r < 8; ++r) o[c][r] *= alpha[r];
            }
        }

        // ---- Transpose P (C-layout f32) -> A-layout bf16 via per-wave LDS ----
        #pragma unroll
        for (int r = 0; r < 8; ++r) {
            const int row = r + 8 * sub;
            Pb[wave][row][l16]      = (__bf16)s0[r];
            Pb[wave][row][16 + l16] = (__bf16)s1[r];
        }
        asm volatile("s_wait_dscnt 0" ::: "memory");

        FragBF pa;
        pa.f.lo = *(const float4*)&Pb[wave][l16][sub * 8];
        pa.f.hi = *(const float4*)&Pb[wave][l16][16 + sub * 8];

        // ---- O += P * V : 16 D-chunks ----
        #pragma unroll
        for (int c = 0; c < 16; ++c) {
            FragBF vb;
            vb.f.lo = *(const float4*)&Vt[buf][c * 16 + l16][sub * 8];
            vb.f.hi = *(const float4*)&Vt[buf][c * 16 + l16][16 + sub * 8];
            o[c] = __builtin_amdgcn_wmma_f32_16x16x32_bf16(false, pa.v, false, vb.v,
                                                           (short)0, o[c], false, false);
        }
        buf ^= 1;
    }

    // ---- Normalize and store ----
    #pragma unroll
    for (int r = 0; r < 8; ++r) {
        const float inv = 1.0f / lrow[r];
        const size_t qrow = (size_t)(qbase + r + 8 * sub);
        #pragma unroll
        for (int c = 0; c < 16; ++c)
            Og[qrow * DD + c * 16 + l16] = o[c][r] * inv;
    }
}

extern "C" void kernel_launch(void* const* d_in, const int* in_sizes, int n_in,
                              void* d_out, int out_size, void* d_ws, size_t ws_size,
                              hipStream_t stream) {
    const float*         Kg = (const float*)d_in[0];
    const float*         Vg = (const float*)d_in[1];
    const float*         Qg = (const float*)d_in[2];
    const unsigned char* Mg = (const unsigned char*)d_in[3];
    float*               Og = (float*)d_out;

    char* ws = (char*)d_ws;
    __bf16*   Kbf = (__bf16*)(ws);                            //  4 MB
    __bf16*   Vtg = (__bf16*)(ws + ((size_t) 4 << 20));       //  4 MB
    __bf16*   Qbf = (__bf16*)(ws + ((size_t) 8 << 20));       //  4 MB
    unsigned* Mw  = (unsigned*)(ws + ((size_t)12 << 20));     //  8 MB

    convert_kernel<<<dim3((NKK * DD) / 256), dim3(256), 0, stream>>>(Kg, Vg, Qg, Kbf, Vtg, Qbf);
    maskpack_kernel<<<dim3((NQQ * NKW) / 256), dim3(256), 0, stream>>>(Mg, Mw);

    fa_wmma_kernel<<<dim3(NQQ / 64), dim3(128), 0, stream>>>(Kbf, Vtg, Qbf, Mw, Og);
}
